// MultiHeadAttention_49117245997325
// MI455X (gfx1250) — matrix operand order repaired and checked by reading, compiled, not run on hardware
//
#include <hip/hip_runtime.h>
#include <hip/hip_bf16.h>

typedef _Float16 half_t;
typedef __attribute__((ext_vector_type(16))) _Float16 v16h;
typedef __attribute__((ext_vector_type(8)))  _Float16 v8h;
typedef __attribute__((ext_vector_type(8)))  float    v8f;
typedef int v4i_t __attribute__((vector_size(16)));   // matches builtin param type

#define D_MODEL   1024
#define NUM_HEADS 16
#define DK        64
#define BB        4
#define SS        2048
#define MTOT      (BB*SS)

static __device__ __forceinline__ v8f wmma_f16(v16h a, v16h b, v8f c) {
  // (neg_a, A, neg_b, B, c_mod, C, reuse_a, reuse_b)
  return __builtin_amdgcn_wmma_f32_16x16x32_f16(false, a, false, b, (short)0, c,
                                                false, false);
}

// ---- CDNA5 async global->LDS copy (ASYNCcnt-tracked), with safe fallback ----
#if defined(__has_builtin)
#  if __has_builtin(__builtin_amdgcn_global_load_async_to_lds_b128)
#    define HAVE_ASYNC_LDS 1
#  endif
#endif
#ifndef HAVE_ASYNC_LDS
#  define HAVE_ASYNC_LDS 0
#endif

static __device__ __forceinline__ void g2l_b128(half_t* lds, const half_t* g) {
#if HAVE_ASYNC_LDS
  // addrspace(1) global ptr: flat address of global memory == its AS1 address.
  // addrspace(3) LDS ptr (32-bit): low 32 bits of the flat LDS address.
  __attribute__((address_space(1))) v4i_t* gp =
      (__attribute__((address_space(1))) v4i_t*)(unsigned long long)(const void*)g;
  __attribute__((address_space(3))) v4i_t* lp =
      (__attribute__((address_space(3))) v4i_t*)(unsigned)(unsigned long long)(void*)lds;
  __builtin_amdgcn_global_load_async_to_lds_b128(gp, lp, 0, 0);
#else
  *(v8h*)lds = *(const v8h*)g;
#endif
}
static __device__ __forceinline__ void g2l_wait() {
#if HAVE_ASYNC_LDS
#  if __has_builtin(__builtin_amdgcn_s_wait_asynccnt)
  __builtin_amdgcn_s_wait_asynccnt(0);
#  else
  asm volatile("s_wait_asynccnt 0" ::: "memory");
#  endif
#endif
}

// ---------------------------------------------------------------------------
// Generic GEMM: C[M,N] = A[M,K] * W^T + bias      (W is [N,K] row-major)
// M=8192, N=K=1024 fixed. A is f32 (AISHALF=0) or f16 (AISHALF=1, async-LDS).
// MODE 0: f32 row-major [M,N]          (final output projection)
// MODE 1: f16 head-split [b,h,s,dk]    (Q, K)
// MODE 2: f16 transposed [b,h,dk,s]    (V^T for the P@V stage)
// Block tile 128(M) x 128(N), 8 waves (4Mx2N), 32x64 per wave, K-step 32.
// Software pipeline: next K-slab global loads overlap the WMMA burst.
// Compile-time AISHALF/MODE -> straight-line inner loop (no uniform branches).
// ---------------------------------------------------------------------------
#define GT_M 128
#define GT_N 128
#define GK   32
#define ASTR 40   // halfs per LDS row (80B: 16B-aligned, conflict-poor)
#define BSTR 40

template<int AISHALF, int MODE>
__global__ __launch_bounds__(256)
void gemm_wmma(const void* __restrict__ Aptr,
               const float* __restrict__ W, const float* __restrict__ bias,
               void* __restrict__ Cptr)
{
  __shared__ __align__(16) half_t As[GT_M * ASTR];
  __shared__ __align__(16) half_t Bs[GT_N * BSTR];

  const int t      = threadIdx.x;
  const int lane   = t & 31;
  const int wave   = t >> 5;
  const int wm     = wave & 3;       // wave M index (0..3)
  const int wn     = wave >> 2;      // wave N index (0..1)
  const int halfid = lane >> 4;
  const int lr     = lane & 15;

  const int m0 = blockIdx.y * GT_M;
  const int n0 = blockIdx.x * GT_N;

  v8f acc[2][4] = {};

  // cooperative-load coordinates: 128 rows x 32 cols, 16 elems/thread
  const int grow = t >> 1;           // 0..127
  const int gcol = (t & 1) * 16;     // 0 / 16

  const float*  Af32 = (const float*)Aptr;
  const half_t* Af16 = (const half_t*)Aptr;

  float4 pa[4], pw[4];
  auto load_w = [&](int kk) {
    const float4* s = (const float4*)(W + (size_t)(n0 + grow) * D_MODEL + kk + gcol);
    pw[0] = s[0]; pw[1] = s[1]; pw[2] = s[2]; pw[3] = s[3];
  };
  auto load_a = [&](int kk) {
    const float4* s = (const float4*)(Af32 + (size_t)(m0 + grow) * D_MODEL + kk + gcol);
    pa[0] = s[0]; pa[1] = s[1]; pa[2] = s[2]; pa[3] = s[3];
  };
  auto cvt_store = [&](half_t* dst, const float4* v) {
    #pragma unroll
    for (int i = 0; i < 4; ++i) {
      half_t* d = dst + i * 4;
      d[0] = (half_t)v[i].x; d[1] = (half_t)v[i].y;
      d[2] = (half_t)v[i].z; d[3] = (half_t)v[i].w;
    }
  };

  load_w(0);
  if (!AISHALF) load_a(0);

  for (int kk = 0; kk < D_MODEL; kk += GK) {
    // --- stage LDS tiles for this K-slab ---
    if (AISHALF) {
      const half_t* g = Af16 + (size_t)(m0 + grow) * D_MODEL + kk + gcol;
      g2l_b128(&As[grow * ASTR + gcol],     g);
      g2l_b128(&As[grow * ASTR + gcol + 8], g + 8);
    } else {
      cvt_store(&As[grow * ASTR + gcol], pa);
    }
    cvt_store(&Bs[grow * BSTR + gcol], pw);
    if (AISHALF) g2l_wait();
    __syncthreads();

    // --- prefetch next K-slab (overlaps the WMMA burst) ---
    if (kk + GK < D_MODEL) {
      load_w(kk + GK);
      if (!AISHALF) load_a(kk + GK);
    }

    // --- fragments + 8 WMMAs ---
    v16h af[2], bf[4];
    #pragma unroll
    for (int tm = 0; tm < 2; ++tm) {
      const int row = wm * 32 + tm * 16 + lr;
      union { v16h v; v8h h2[2]; } u;
      u.h2[0] = *(const v8h*)&As[row * ASTR + 8 * halfid];        // K = 8h+0..7
      u.h2[1] = *(const v8h*)&As[row * ASTR + 16 + 8 * halfid];   // K = 16+8h+0..7
      af[tm] = u.v;
    }
    #pragma unroll
    for (int tn = 0; tn < 4; ++tn) {
      const int col = wn * 64 + tn * 16 + lr;
      union { v16h v; v8h h2[2]; } u;
      u.h2[0] = *(const v8h*)&Bs[col * BSTR + 8 * halfid];        // K = 8h+0..7
      u.h2[1] = *(const v8h*)&Bs[col * BSTR + 16 + 8 * halfid];   // K = 16+8h+0..7
      bf[tn] = u.v;
    }
    #pragma unroll
    for (int tm = 0; tm < 2; ++tm)
      #pragma unroll
      for (int tn = 0; tn < 4; ++tn)
        acc[tm][tn] = wmma_f16(af[tm], bf[tn], acc[tm][tn]);

    __syncthreads();
  }

  // --- epilogue: bias + store ---
  #pragma unroll
  for (int tm = 0; tm < 2; ++tm) {
    #pragma unroll
    for (int tn = 0; tn < 4; ++tn) {
      const int col = n0 + wn * 64 + tn * 16 + lr;
      const float bv = bias[col];
      #pragma unroll
      for (int r = 0; r < 8; ++r) {
        const int row = m0 + wm * 32 + tm * 16 + 8 * halfid + r;
        const float val = acc[tm][tn][r] + bv;
        if (MODE == 0) {
          ((float*)Cptr)[(size_t)row * D_MODEL + col] = val;
        } else {
          const int b = row >> 11, s = row & (SS - 1);
          const int h = col >> 6,  d = col & (DK - 1);
          const half_t hv = (half_t)val;
          if (MODE == 1)
            ((half_t*)Cptr)[((((size_t)b * NUM_HEADS + h) * SS + s) << 6) + d] = hv;
          else
            ((half_t*)Cptr)[(((size_t)b * NUM_HEADS + h) * DK + d) * SS + s] = hv;
        }
      }
    }
  }
}

// ---------------------------------------------------------------------------
// Flash attention: per wave 16 query rows, streaming softmax over 2048 keys
// in chunks of 32.  Q,K in [b,h,s,64] f16;  V^T in [b,h,64,s] f16.
// K/V chunks are staged once per block into LDS (async global->LDS), shared
// by all 8 waves.  Output AO in [b,s,h,dk] f16 (== [b,s,d_model]).
// ---------------------------------------------------------------------------
#define PSTR 40
#define KSTR 72   // 32 rows x 64 halfs, padded (144B rows, 16B aligned)
#define VSTR 40   // 64 rows x 32 halfs, padded

__global__ __launch_bounds__(256)
void attn_flash(const half_t* __restrict__ Qh, const half_t* __restrict__ Kh,
                const half_t* __restrict__ VT, half_t* __restrict__ AO)
{
  __shared__ __align__(16) half_t Ks[32 * KSTR];
  __shared__ __align__(16) half_t Vs[64 * VSTR];
  __shared__ __align__(16) half_t P[8 * 16 * PSTR];

  const int t      = threadIdx.x;
  const int lane   = t & 31;
  const int wave   = t >> 5;
  const int halfid = lane >> 4;
  const int lr     = lane & 15;

  const int bh = blockIdx.y;              // 0..63
  const int b  = bh >> 4, h = bh & 15;
  const int q0 = blockIdx.x * 128 + wave * 16;
  const size_t head = (size_t)bh * SS * DK;
  const float scale = 0.125f;             // 1/sqrt(64)

  // cooperative K/V staging coordinates (one b128 each per thread per chunk)
  const int krow = t >> 3, kseg = t & 7;  // 32 rows x 8 segs
  const int vrow = t >> 2, vseg = t & 3;  // 64 rows x 4 segs

  // Q fragments (A layout), loaded once per wave
  v16h qf[2];
  {
    const size_t qb = head + (size_t)(q0 + lr) * DK;
    #pragma unroll
    for (int ks = 0; ks < 2; ++ks) {
      union { v16h v; v8h h2[2]; } u;
      u.h2[0] = *(const v8h*)&Qh[qb + ks * 32 + 8 * halfid];
      u.h2[1] = *(const v8h*)&Qh[qb + ks * 32 + 16 + 8 * halfid];
      qf[ks] = u.v;
    }
  }

  v8f o[4] = {};
  float mrow[8], lrow[8];
  #pragma unroll
  for (int r = 0; r < 8; ++r) { mrow[r] = -1e30f; lrow[r] = 0.0f; }

  half_t* Pw = &P[wave * 16 * PSTR];      // wave-private P staging tile

  for (int c = 0; c < SS; c += 32) {
    // ---- stage K (32x64) and V^T (64x32) chunks into LDS, async ----
    g2l_b128(&Ks[krow * KSTR + kseg * 8],
             Kh + head + (size_t)(c + krow) * DK + kseg * 8);
    g2l_b128(&Vs[vrow * VSTR + vseg * 8],
             VT + head + (size_t)vrow * SS + c + vseg * 8);
    g2l_wait();
    __syncthreads();

    // ---- scores: S = Q @ K^T (16 x 32) ----
    v8f sc[2] = {};
    #pragma unroll
    for (int tn = 0; tn < 2; ++tn) {
      #pragma unroll
      for (int ks = 0; ks < 2; ++ks) {
        union { v16h v; v8h h2[2]; } u;
        u.h2[0] = *(const v8h*)&Ks[(tn * 16 + lr) * KSTR + ks * 32 + 8 * halfid];
        u.h2[1] = *(const v8h*)&Ks[(tn * 16 + lr) * KSTR + ks * 32 + 16 + 8 * halfid];
        sc[tn] = wmma_f16(qf[ks], u.v, sc[tn]);
      }
    }

    // ---- online softmax (row = r + 8*halfid, 16 lanes per row-group) ----
    #pragma unroll
    for (int r = 0; r < 8; ++r) {
      const float x0 = sc[0][r] * scale;
      const float x1 = sc[1][r] * scale;
      float mx = fmaxf(x0, x1);
      mx = fmaxf(mx, __shfl_xor(mx, 1, 32));
      mx = fmaxf(mx, __shfl_xor(mx, 2, 32));
      mx = fmaxf(mx, __shfl_xor(mx, 4, 32));
      mx = fmaxf(mx, __shfl_xor(mx, 8, 32));
      const float mnew  = fmaxf(mrow[r], mx);
      const float alpha = __expf(mrow[r] - mnew);
      const float p0 = __expf(x0 - mnew);
      const float p1 = __expf(x1 - mnew);
      float ps = p0 + p1;
      ps += __shfl_xor(ps, 1, 32);
      ps += __shfl_xor(ps, 2, 32);
      ps += __shfl_xor(ps, 4, 32);
      ps += __shfl_xor(ps, 8, 32);
      lrow[r] = lrow[r] * alpha + ps;
      mrow[r] = mnew;
      o[0][r] *= alpha; o[1][r] *= alpha; o[2][r] *= alpha; o[3][r] *= alpha;
      // C-layout -> LDS (row-major 16x32 P tile)
      Pw[(8 * halfid + r) * PSTR + lr]      = (half_t)p0;
      Pw[(8 * halfid + r) * PSTR + 16 + lr] = (half_t)p1;
    }
    // order same-wave LDS store -> ds_load (CDNA5 split DS counter)
    asm volatile("s_wait_dscnt 0" ::: "memory");

    // ---- re-fragment P into A layout ----
    union { v16h v; v8h h2[2]; } pu;
    pu.h2[0] = *(const v8h*)&Pw[lr * PSTR + 8 * halfid];
    pu.h2[1] = *(const v8h*)&Pw[lr * PSTR + 16 + 8 * halfid];

    // ---- O += P @ V  (V^T fragments from LDS) ----
    #pragma unroll
    for (int tv = 0; tv < 4; ++tv) {
      union { v16h v; v8h h2[2]; } vu;
      vu.h2[0] = *(const v8h*)&Vs[(tv * 16 + lr) * VSTR + 8 * halfid];
      vu.h2[1] = *(const v8h*)&Vs[(tv * 16 + lr) * VSTR + 16 + 8 * halfid];
      o[tv] = wmma_f16(pu.v, vu.v, o[tv]);
    }

    __syncthreads();   // all waves done with Ks/Vs before next chunk staging
  }

  // ---- normalize + store f16 [b,s,h,dk] ----
  #pragma unroll
  for (int r = 0; r < 8; ++r) {
    const float inv  = 1.0f / lrow[r];
    const int   qrow = q0 + 8 * halfid + r;
    const size_t ob  = ((size_t)b * SS + qrow) * D_MODEL + h * DK + lr;
    #pragma unroll
    for (int tv = 0; tv < 4; ++tv)
      AO[ob + tv * 16] = (half_t)(o[tv][r] * inv);
  }
}

// ---------------------------------------------------------------------------
// Host-side launch.  Inputs (dict order):
// 0 query, 1 key, 2 value, 3 Wq, 4 bq, 5 Wk, 6 bk, 7 Wv, 8 bv, 9 Wo, 10 bo
// Workspace: Qh | Kh | VT | AO, each B*H*S*DK f16 = 16 MB (64 MB total).
// ---------------------------------------------------------------------------
extern "C" void kernel_launch(void* const* d_in, const int* in_sizes, int n_in,
                              void* d_out, int out_size, void* d_ws, size_t ws_size,
                              hipStream_t stream)
{
  const float* q  = (const float*)d_in[0];
  const float* k  = (const float*)d_in[1];
  const float* v  = (const float*)d_in[2];
  const float* Wq = (const float*)d_in[3];
  const float* bq = (const float*)d_in[4];
  const float* Wk = (const float*)d_in[5];
  const float* bk = (const float*)d_in[6];
  const float* Wv = (const float*)d_in[7];
  const float* bv = (const float*)d_in[8];
  const float* Wo = (const float*)d_in[9];
  const float* bo = (const float*)d_in[10];

  const size_t perBuf = (size_t)BB * NUM_HEADS * SS * DK;   // 8Mi halfs
  half_t* Qh = (half_t*)d_ws;
  half_t* Kh = Qh + perBuf;
  half_t* VT = Kh + perBuf;
  half_t* AO = VT + perBuf;

  dim3 gg(D_MODEL / GT_N, MTOT / GT_M);   // (8, 64)
  dim3 bb(256);

  gemm_wmma<0,1><<<gg, bb, 0, stream>>>(q, Wq, bq, Qh);
  gemm_wmma<0,1><<<gg, bb, 0, stream>>>(k, Wk, bk, Kh);
  gemm_wmma<0,2><<<gg, bb, 0, stream>>>(v, Wv, bv, VT);

  attn_flash<<<dim3(SS / 128, BB * NUM_HEADS), bb, 0, stream>>>(Qh, Kh, VT, AO);

  gemm_wmma<1,0><<<gg, bb, 0, stream>>>(AO, Wo, bo, d_out);
}
